// RPEAttention_17197049053734
// MI455X (gfx1250) — matrix operand order repaired and checked by reading, not compiled, not run on hardware
//
#include <hip/hip_runtime.h>
#include <hip/hip_bf16.h>
#include <math.h>

#define BDIM 4
#define SEQ 2048
#define HEADS 8
#define DH 32
#define CDIM 256
#define QKV_N 768

typedef __attribute__((ext_vector_type(16))) __bf16 bf16x16;
typedef __attribute__((ext_vector_type(8)))  float  f32x8;
typedef __attribute__((ext_vector_type(4)))  float  f32x4;

union TileU { bf16x16 v; uint4 q[2]; };

__device__ __forceinline__ unsigned short f2bf(float f) {
  unsigned int u = __float_as_uint(f);
  u += 0x7FFFu + ((u >> 16) & 1u);       // round-to-nearest-even
  return (unsigned short)(u >> 16);
}

// 16x32 bf16 A tile from row-major [ld] source.
// lane m (0-15): row m, K=0..7 & 16..23 ; lane m+16: row m, K=8..15 & 24..31
__device__ __forceinline__ bf16x16 load_A16x32(const unsigned short* base, int ld, int lane) {
  const int m  = lane & 15;
  const int hi = (lane >> 4) & 1;
  const unsigned short* p = base + m * ld + hi * 8;
  TileU u;
  u.q[0] = *(const uint4*)(p);
  u.q[1] = *(const uint4*)(p + 16);
  return u.v;
}

// 32x16 bf16 B tile; source row n holds K-contiguous data for output column n.
// lane n (0-15): col n, K=0..7 & 16..23 ; lane n+16: col n, K=8..15 & 24..31
__device__ __forceinline__ bf16x16 load_B32x16(const unsigned short* base, int ld, int lane) {
  const int n  = lane & 15;
  const int hi = (lane >> 4) & 1;
  const unsigned short* p = base + n * ld + hi * 8;
  TileU u;
  u.q[0] = *(const uint4*)(p);
  u.q[1] = *(const uint4*)(p + 16);
  return u.v;
}

__device__ __forceinline__ f32x8 wmma_bf16(bf16x16 a, bf16x16 b, f32x8 c) {
  return __builtin_amdgcn_wmma_f32_16x16x32_bf16(false, a, false, b, (short)0, c, false, false);
}

// ---------------- kernel 0: f32 -> bf16 conversions ----------------
__global__ __launch_bounds__(256) void k_convert(const float* __restrict__ x,
                                                 const float* __restrict__ wqkv,
                                                 const float* __restrict__ wout,
                                                 unsigned short* __restrict__ xb,
                                                 unsigned short* __restrict__ wqkvb,
                                                 unsigned short* __restrict__ woutb) {
  const int i = blockIdx.x * blockDim.x + threadIdx.x;
  const int stride = gridDim.x * blockDim.x;
  const int NX = BDIM * SEQ * CDIM;   // 2,097,152
  const int NW = QKV_N * CDIM;        //   196,608
  const int NO = CDIM * CDIM;         //    65,536
  for (int t = i; t < NX; t += stride) xb[t]    = f2bf(x[t]);
  for (int t = i; t < NW; t += stride) wqkvb[t] = f2bf(wqkv[t]);
  for (int t = i; t < NO; t += stride) woutb[t] = f2bf(wout[t]);
}

// ---------------- kernel 1: QKV projection + scatter ----------------
// y[m, c] = sum_k x[m,k]*W[c,k] + b[c], then scatter into
// Q [B,H,N,Dh] bf16, K [B,H,N,Dh] bf16, Vt [B,H,Dh,N] bf16
__global__ __launch_bounds__(128) void k_qkv(const unsigned short* __restrict__ xb,
                                             const unsigned short* __restrict__ wb,
                                             const float* __restrict__ bias,
                                             unsigned short* __restrict__ qws,
                                             unsigned short* __restrict__ kws,
                                             unsigned short* __restrict__ vtws) {
  const int lane = threadIdx.x & 31;
  const int wid  = threadIdx.x >> 5;
  const int tile = blockIdx.x * 4 + wid;          // 24576 tiles
  const int nt = tile % (QKV_N / 16);
  const int mt = tile / (QKV_N / 16);
  const int m0 = mt * 16, n0 = nt * 16;

  f32x8 acc = {};
#pragma unroll
  for (int k0 = 0; k0 < CDIM; k0 += 32) {
    bf16x16 a = load_A16x32(xb + m0 * CDIM + k0, CDIM, lane);
    bf16x16 b = load_B32x16(wb + n0 * CDIM + k0, CDIM, lane);
    acc = wmma_bf16(a, b, acc);
  }

  const int c   = n0 + (lane & 15);
  const int hi  = (lane >> 4) & 1;
  const float bv = bias[c];
  const int mat = c >> 8;          // 0=Q, 1=K, 2=V
  const int cc  = c & 255;
  const int h   = cc >> 5;
  const int d   = cc & 31;
#pragma unroll
  for (int r = 0; r < 8; ++r) {
    const int m  = m0 + r + hi * 8;
    const int bb = m >> 11;                        // / 2048
    const int n  = m & (SEQ - 1);
    const unsigned short val = f2bf(acc[r] + bv);
    const int bh = bb * HEADS + h;
    if (mat == 0)      qws[(bh * SEQ + n) * DH + d] = val;
    else if (mat == 1) kws[(bh * SEQ + n) * DH + d] = val;
    else               vtws[(bh * DH + d) * SEQ + n] = val;
  }
}

// ---------------- kernel 2: flash attention, S^T layout softmax ----------------
// S^T = K_tile(A) x Q_tile(B): lane owns one query; its 16 key-logits sit in
// its own VGPRs -> lane-local softmax reductions + one xor-16 merge.
// rpe is streamed with non-temporal loads so the once-read 536MB stream does
// not evict the K/V/Q working set from L2.
__global__ __launch_bounds__(128) void k_attn(const unsigned short* __restrict__ qws,
                                              const unsigned short* __restrict__ kws,
                                              const unsigned short* __restrict__ vtws,
                                              const float* __restrict__ rpe,
                                              const int* __restrict__ mask,
                                              unsigned short* __restrict__ ao) {
  __shared__ __align__(16) unsigned short Plds[4][16][32];
  const int lane = threadIdx.x & 31;
  const int wid  = threadIdx.x >> 5;
  const int tile = blockIdx.x * 4 + wid;          // 4096 tiles
  const int qt = tile & 127;
  const int h  = (tile >> 7) & 7;
  const int b  = tile >> 10;
  const int q0 = qt * 16;
  const int bh = b * HEADS + h;
  const float scale = 0.17677669529663687f;       // 1/sqrt(32)
  const int ql  = lane & 15;                      // this lane's query (and dim col)
  const int hi  = (lane >> 4) & 1;

  // Q as B-matrix (lane = query, dims K-contiguous), loaded once
  const bf16x16 bq = load_B32x16(qws + (bh * SEQ + q0) * DH, DH, lane);

  float m_run = -__builtin_inff();
  float l_run = 0.f;
  f32x8 o0 = {}, o1 = {};

  // this lane's rpe row: rpe[b,h, q0+ql, :]
  const float* rp_row = rpe + ((size_t)bh * SEQ + (size_t)(q0 + ql)) * SEQ;
  const int mbase = b * SEQ;

  for (int k0 = 0; k0 < SEQ; k0 += 32) {
    // K tiles as A-matrix (rows = keys)
    bf16x16 ak0 = load_A16x32(kws + (bh * SEQ + k0) * DH, DH, lane);
    bf16x16 ak1 = load_A16x32(kws + (bh * SEQ + k0 + 16) * DH, DH, lane);
    f32x8 z = {};
    f32x8 s0 = wmma_bf16(ak0, bq, z);   // S^T keys k0+{0..15}  (this lane: keys hi*8 + 0..7)
    f32x8 s1 = wmma_bf16(ak1, bq, z);   // S^T keys k0+16+{0..15}

    // rpe for this lane's 16 (query,key) logits: 4 contiguous NT float4 loads
    const f32x4 rp0a = __builtin_nontemporal_load((const f32x4*)(rp_row + k0 + hi * 8));
    const f32x4 rp0b = __builtin_nontemporal_load((const f32x4*)(rp_row + k0 + hi * 8 + 4));
    const f32x4 rp1a = __builtin_nontemporal_load((const f32x4*)(rp_row + k0 + 16 + hi * 8));
    const f32x4 rp1b = __builtin_nontemporal_load((const f32x4*)(rp_row + k0 + 16 + hi * 8 + 4));
    if (k0 + 32 < SEQ) __builtin_prefetch(rp_row + k0 + 32 + hi * 8, 0, 1);

    // key padding: one lane-per-key load, wave-uniform ballot
    const unsigned long long bal64 = __ballot(mask[mbase + k0 + lane] != 0);
    const unsigned int bal  = (unsigned int)bal64;
    const unsigned int bal0 = bal >> (hi * 8);          // bits for s0 keys
    const unsigned int bal1 = bal >> (16 + hi * 8);     // bits for s1 keys

    float t0[8], t1[8];
#pragma unroll
    for (int r = 0; r < 4; ++r) {
      t0[r]     = s0[r]     * scale + rp0a[r];
      t0[4 + r] = s0[4 + r] * scale + rp0b[r];
      t1[r]     = s1[r]     * scale + rp1a[r];
      t1[4 + r] = s1[4 + r] * scale + rp1b[r];
    }

    // tile max: lane-local over 16 logits, then merge the other half's keys
    float mt = t0[0];
#pragma unroll
    for (int r = 1; r < 8; ++r) mt = fmaxf(mt, t0[r]);
#pragma unroll
    for (int r = 0; r < 8; ++r) mt = fmaxf(mt, t1[r]);
    mt = fmaxf(mt, __shfl_xor(mt, 16, 32));

    const float mn = fmaxf(m_run, mt);
    const float alpha = __expf(m_run - mn);
    m_run = mn;

    // probabilities; zero out padded keys after exp (mask in max is harmless)
    float sum = 0.f;
#pragma unroll
    for (int r = 0; r < 8; ++r) {
      float p = __expf(t0[r] - mn);
      p = ((bal0 >> r) & 1u) ? 0.f : p;
      t0[r] = p; sum += p;
    }
#pragma unroll
    for (int r = 0; r < 8; ++r) {
      float p = __expf(t1[r] - mn);
      p = ((bal1 >> r) & 1u) ? 0.f : p;
      t1[r] = p; sum += p;
    }
    sum += __shfl_xor(sum, 16, 32);
    l_run = l_run * alpha + sum;

    // store P row-major [q][key] (contiguous 8-key runs per lane)
    {
      unsigned short* prow = &Plds[wid][ql][hi * 8];
#pragma unroll
      for (int r = 0; r < 8; ++r) prow[r]      = f2bf(t0[r]);
#pragma unroll
      for (int r = 0; r < 8; ++r) prow[16 + r] = f2bf(t1[r]);
    }
    __builtin_amdgcn_wave_barrier();   // DS ops are wave-in-order; fence the scheduler
    const bf16x16 ap = load_A16x32(&Plds[wid][0][0], 32, lane);
    __builtin_amdgcn_wave_barrier();

    // rescale O rows (row = query r + hi*8) by alpha broadcast from owning lane
#pragma unroll
    for (int r = 0; r < 8; ++r) {
      const float a_r = __shfl(alpha, r + hi * 8, 32);
      o0[r] *= a_r;
      o1[r] *= a_r;
    }

    bf16x16 bv0 = load_B32x16(vtws + (bh * DH + 0)  * SEQ + k0, SEQ, lane);
    bf16x16 bv1 = load_B32x16(vtws + (bh * DH + 16) * SEQ + k0, SEQ, lane);
    o0 = wmma_bf16(ap, bv0, o0);
    o1 = wmma_bf16(ap, bv1, o1);
  }

  // epilogue: normalize (1/l broadcast per row) and write attn_out [B, N, C] bf16
  const float linv = 1.f / l_run;
#pragma unroll
  for (int r = 0; r < 8; ++r) {
    const float inv = __shfl(linv, r + hi * 8, 32);
    const int q = q0 + r + hi * 8;
    unsigned short* dst = ao + ((b * SEQ + q) * CDIM) + h * DH;
    dst[ql]      = f2bf(o0[r] * inv);
    dst[16 + ql] = f2bf(o1[r] * inv);
  }
}

// ---------------- kernel 3: output projection ----------------
__global__ __launch_bounds__(128) void k_out(const unsigned short* __restrict__ ab,
                                             const unsigned short* __restrict__ wb,
                                             const float* __restrict__ bias,
                                             float* __restrict__ out) {
  const int lane = threadIdx.x & 31;
  const int wid  = threadIdx.x >> 5;
  const int tile = blockIdx.x * 4 + wid;   // 8192 tiles
  const int nt = tile & 15;
  const int mt = tile >> 4;
  const int m0 = mt * 16, n0 = nt * 16;

  f32x8 acc = {};
#pragma unroll
  for (int k0 = 0; k0 < CDIM; k0 += 32) {
    bf16x16 a = load_A16x32(ab + m0 * CDIM + k0, CDIM, lane);
    bf16x16 b = load_B32x16(wb + n0 * CDIM + k0, CDIM, lane);
    acc = wmma_bf16(a, b, acc);
  }
  const int c  = n0 + (lane & 15);
  const int hi = (lane >> 4) & 1;
  const float bv = bias[c];
#pragma unroll
  for (int r = 0; r < 8; ++r)
    out[(m0 + r + hi * 8) * CDIM + c] = acc[r] + bv;
}

extern "C" void kernel_launch(void* const* d_in, const int* in_sizes, int n_in,
                              void* d_out, int out_size, void* d_ws, size_t ws_size,
                              hipStream_t stream) {
  const float* x      = (const float*)d_in[0];
  const float* rpe    = (const float*)d_in[1];
  const int*   mask   = (const int*)d_in[2];
  const float* qkv_w  = (const float*)d_in[3];
  const float* qkv_b  = (const float*)d_in[4];
  const float* out_w  = (const float*)d_in[5];
  const float* out_b  = (const float*)d_in[6];
  float* out = (float*)d_out;

  unsigned short* wsp   = (unsigned short*)d_ws;
  unsigned short* xb    = wsp;                 // 2,097,152
  unsigned short* wqkvb = xb    + 2097152;     //   196,608
  unsigned short* woutb = wqkvb + 196608;      //    65,536
  unsigned short* qws   = woutb + 65536;       // 2,097,152
  unsigned short* kws   = qws   + 2097152;     // 2,097,152
  unsigned short* vtws  = kws   + 2097152;     // 2,097,152
  unsigned short* ao    = vtws  + 2097152;     // 2,097,152  (~20.5 MB total)

  hipLaunchKernelGGL(k_convert, dim3(2048), dim3(256), 0, stream,
                     x, qkv_w, out_w, xb, wqkvb, woutb);
  hipLaunchKernelGGL(k_qkv, dim3(6144), dim3(128), 0, stream,
                     xb, wqkvb, qkv_b, qws, kws, vtws);
  hipLaunchKernelGGL(k_attn, dim3(1024), dim3(128), 0, stream,
                     qws, kws, vtws, rpe, mask, ao);
  hipLaunchKernelGGL(k_out, dim3(2048), dim3(128), 0, stream,
                     ao, woutb, out_b, out);
}
